// HAN_2018634629254
// MI455X (gfx1250) — matrix operand order repaired and checked by reading, not compiled, not run on hardware
//
#include <hip/hip_runtime.h>
#include <hip/hip_bf16.h>

typedef __attribute__((ext_vector_type(16))) _Float16 v16h;
typedef __attribute__((ext_vector_type(8)))  float    v8f;

#define HID   256
#define NHEAD 8
#define KMAX  256
#define BS_STRIDE (KMAX + 16)   // halves; pad to spread LDS banks (136 dwords/row)

// ======================================================================
// WMMA GEMM (f32 inputs converted to f16 in registers):
//   acc = A[M,K] * B[K,N]   via v_wmma_f32_16x16x32_f16
// mode 0 : C[m,n] = acc + bias[n]
// mode 1 : colsum[n] += sum_m tanh(acc + bias[n])   (semantic attention)
// REQUIRES: M % 16 == 0, K % 32 == 0, N % 64 == 0 (true for all call sites).
// Block = 8 waves sharing one 64-col group; B panel staged in LDS (f16,
// col-major) so each lane's B fragment is 2x ds_load_b128. A is read with
// float4 -> global_load_b128. All 4 B fragments are prefetched before the
// WMMA burst so the ds waits pipeline instead of draining per-WMMA.
// ======================================================================
__global__ __launch_bounds__(256)
void gemm_wmma_f16(const float* __restrict__ A, const float* __restrict__ B,
                   const float* __restrict__ bias, float* __restrict__ C,
                   float* __restrict__ colsum, int M, int K, int N, int mode)
{
  __shared__ _Float16 Bs[64 * BS_STRIDE];

  const int tid  = threadIdx.x;
  const int lane = tid & 31;
  const int wave = tid >> 5;
  const int colGroups = N >> 6;                 // 64-wide column groups
  const int rowTiles  = M >> 4;
  const int cg  = blockIdx.x % colGroups;
  const int brt = blockIdx.x / colGroups;       // block of 8 row tiles

  // ---- cooperative stage: B[k][cg*64+c] -> Bs[c][k] as f16 (coalesced reads)
  {
    const int c  = tid & 63;
    const int k0 = tid >> 6;
    for (int k = k0; k < K; k += 4)
      Bs[c * BS_STRIDE + k] = (_Float16)B[(size_t)k * N + cg * 64 + c];
  }
  __syncthreads();

  const int rowTile = brt * 8 + wave;
  if (rowTile >= rowTiles) return;              // whole wave exits together

  const int r   = rowTile * 16 + (lane & 15);   // A row for this lane
  const int kh  = lane >> 4;                    // K-half selector
  const int nlo = lane & 15;
  const float* Arow = A + (size_t)r * K;

  v8f acc[4];
#pragma unroll
  for (int t = 0; t < 4; ++t)
#pragma unroll
    for (int i = 0; i < 8; ++i) acc[t][i] = 0.0f;

  for (int kb = 0; kb < K; kb += 32) {
    // A fragment 16x32 f16: element j -> K=kb+16*kh+j
    const float4* Alo = (const float4*)(Arow + kb + kh * 16);
    const float4* Ahi = (const float4*)(Arow + kb + kh * 16 + 8);
    const float4 a0 = Alo[0], a1 = Alo[1], a2 = Ahi[0], a3 = Ahi[1];

    // Prefetch all 4 B fragments (2x ds_load_b128 each) before the WMMA burst.
    const _Float16* bbase = &Bs[nlo * BS_STRIDE + kb + kh * 16];
    const v16h b0 = *(const v16h*)(bbase);
    const v16h b1 = *(const v16h*)(bbase + 16 * BS_STRIDE);
    const v16h b2 = *(const v16h*)(bbase + 32 * BS_STRIDE);
    const v16h b3 = *(const v16h*)(bbase + 48 * BS_STRIDE);

    v16h a;
    a[0]=(_Float16)a0.x;  a[1]=(_Float16)a0.y;  a[2]=(_Float16)a0.z;  a[3]=(_Float16)a0.w;
    a[4]=(_Float16)a1.x;  a[5]=(_Float16)a1.y;  a[6]=(_Float16)a1.z;  a[7]=(_Float16)a1.w;
    a[8]=(_Float16)a2.x;  a[9]=(_Float16)a2.y;  a[10]=(_Float16)a2.z; a[11]=(_Float16)a2.w;
    a[12]=(_Float16)a3.x; a[13]=(_Float16)a3.y; a[14]=(_Float16)a3.z; a[15]=(_Float16)a3.w;

    acc[0] = __builtin_amdgcn_wmma_f32_16x16x32_f16(false, a, false, b0, (short)0, acc[0], false, false);
    acc[1] = __builtin_amdgcn_wmma_f32_16x16x32_f16(false, a, false, b1, (short)0, acc[1], false, false);
    acc[2] = __builtin_amdgcn_wmma_f32_16x16x32_f16(false, a, false, b2, (short)0, acc[2], false, false);
    acc[3] = __builtin_amdgcn_wmma_f32_16x16x32_f16(false, a, false, b3, (short)0, acc[3], false, false);
  }

  if (mode == 0) {
#pragma unroll
    for (int t = 0; t < 4; ++t) {
      const int c = cg * 64 + t * 16 + nlo;
      const float bv = bias ? bias[c] : 0.0f;
#pragma unroll
      for (int i = 0; i < 8; ++i) {
        const int m = rowTile * 16 + kh * 8 + i;   // C/D layout: row = 8*kh + i
        C[(size_t)m * N + c] = acc[t][i] + bv;
      }
    }
  } else {
#pragma unroll
    for (int t = 0; t < 4; ++t) {
      const int c = cg * 64 + t * 16 + nlo;
      const float bv = bias[c];
      float s = 0.0f;
#pragma unroll
      for (int i = 0; i < 8; ++i) s += tanhf(acc[t][i] + bv);
      atomicAdd(&colsum[c], s);
    }
  }
}

// ---------------- per-(node,head) attention logits: out[n,h] = <h[n,h,:], att[h,:]> ----
__global__ void att_reduce(const float* __restrict__ h, const float* __restrict__ att,
                           float* __restrict__ out, int N)
{
  int idx = blockIdx.x * blockDim.x + threadIdx.x;
  if (idx >= N * NHEAD) return;
  int n = idx >> 3, hd = idx & 7;
  const float4* hp = (const float4*)(h + (size_t)n * HID + hd * 32);
  const float4* ap = (const float4*)(att + hd * 32);
  float s = 0.0f;
#pragma unroll
  for (int d = 0; d < 8; ++d) {
    float4 hv = hp[d], av = ap[d];
    s += hv.x * av.x + hv.y * av.y + hv.z * av.z + hv.w * av.w;
  }
  out[idx] = s;
}

// Order-preserving float<->uint map for atomicMax-based segment max
__device__ __forceinline__ unsigned fkey(float v) {
  unsigned u = __float_as_uint(v);
  return (u & 0x80000000u) ? ~u : (u | 0x80000000u);
}
__device__ __forceinline__ float funkey(unsigned u) {
  return (u & 0x80000000u) ? __uint_as_float(u & 0x7fffffffu) : __uint_as_float(~u);
}

// ---------------- edge pass 1: leaky-relu logits + segment max --------------------------
__global__ void edge_logits(const int* __restrict__ ei, const float* __restrict__ asrc,
                            const float* __restrict__ adst, float* __restrict__ ebuf,
                            unsigned* __restrict__ segmax, int E)
{
  int idx = blockIdx.x * blockDim.x + threadIdx.x;
  if (idx >= E * NHEAD) return;
  int e = idx >> 3, hd = idx & 7;
  int s = ei[e], d = ei[E + e];
  float v = asrc[s * NHEAD + hd] + adst[d * NHEAD + hd];
  v = v > 0.0f ? v : 0.2f * v;                  // leaky_relu(0.2)
  ebuf[idx] = v;
  atomicMax(&segmax[d * NHEAD + hd], fkey(v));
}

// ---------------- edge pass 2: exp(e - max) + segment denominator ----------------------
__global__ void edge_exp(const int* __restrict__ ei, float* __restrict__ ebuf,
                         const unsigned* __restrict__ segmax, float* __restrict__ denom, int E)
{
  int idx = blockIdx.x * blockDim.x + threadIdx.x;
  if (idx >= E * NHEAD) return;
  int e = idx >> 3, hd = idx & 7;
  int d = ei[E + e];
  float mx = funkey(segmax[d * NHEAD + hd]);
  float ee = __expf(ebuf[idx] - mx);
  ebuf[idx] = ee;
  atomicAdd(&denom[d * NHEAD + hd], ee);
}

// ---------------- edge pass 3: scatter weighted messages -------------------------------
__global__ void edge_msg(const int* __restrict__ ei, const float* __restrict__ ebuf,
                         const float* __restrict__ denom, const float* __restrict__ h,
                         float* __restrict__ agg, int E)
{
  long long idx = (long long)blockIdx.x * blockDim.x + threadIdx.x;
  if (idx >= (long long)E * HID) return;
  int e = (int)(idx >> 8), c = (int)(idx & 255), hd = c >> 5;
  int s = ei[e], d = ei[E + e];
  float alpha = ebuf[e * NHEAD + hd] / (denom[d * NHEAD + hd] + 1e-16f);
  atomicAdd(&agg[(size_t)d * HID + c], h[(size_t)s * HID + c] * alpha);
}

__global__ void relu_ip(float* __restrict__ x, int n)
{
  int i = blockIdx.x * blockDim.x + threadIdx.x;
  if (i < n) x[i] = fmaxf(x[i], 0.0f);
}

// semantic softmax over the two paper relations (author/term have R=1 -> identity)
__global__ void semantic_score(const float* __restrict__ colsum, const float* __restrict__ q,
                               float invN, float* __restrict__ attn)
{
  if (threadIdx.x == 0 && blockIdx.x == 0) {
    float s0 = 0.0f, s1 = 0.0f;
    for (int c = 0; c < HID; ++c) {
      s0 += colsum[c]       * invN * q[c];
      s1 += colsum[HID + c] * invN * q[c];
    }
    float m = fmaxf(s0, s1);
    float e0 = __expf(s0 - m), e1 = __expf(s1 - m);
    float inv = 1.0f / (e0 + e1);
    attn[0] = e0 * inv;
    attn[1] = e1 * inv;
  }
}

__global__ void combine2(float* __restrict__ a, const float* __restrict__ b,
                         const float* __restrict__ attn, int n)
{
  int i = blockIdx.x * blockDim.x + threadIdx.x;
  if (i < n) a[i] = attn[0] * a[i] + attn[1] * b[i];
}

// final classification head: [40000,256] @ [256,10] + b
__global__ void head_gemm(const float* __restrict__ x, const float* __restrict__ w,
                          const float* __restrict__ b, float* __restrict__ out, int M)
{
  int idx = blockIdx.x * blockDim.x + threadIdx.x;
  if (idx >= M * 10) return;
  int n = idx / 10, o = idx % 10;
  const float* xp = x + (size_t)n * HID;
  float s = b[o];
#pragma unroll 8
  for (int c = 0; c < HID; ++c) s += xp[c] * w[c * 10 + o];
  out[idx] = s;
}

// ======================================================================
extern "C" void kernel_launch(void* const* d_in, const int* in_sizes, int n_in,
                              void* d_out, int out_size, void* d_ws, size_t ws_size,
                              hipStream_t stream)
{
  (void)in_sizes; (void)out_size; (void)ws_size;
  const int NA = 40000, NP = 80000, NT = 20000, E = 200000;

  const float* x0[3] = {(const float*)d_in[0], (const float*)d_in[1], (const float*)d_in[2]};
  const int*   ei[4] = {(const int*)d_in[3], (const int*)d_in[4],
                        (const int*)d_in[5], (const int*)d_in[6]};

  // Parameter leaves in JAX pytree flatten order (dict keys sorted):
  // head_b, head_w, then per layer:
  //   att_dst[AP,PA,PT,TP], att_src[AP,PA,PT,TP], k_b, k_w,
  //   proj author{b,w}, paper{b,w}, term{b,w}, q      (17 leaves/layer)
  const float* P[53];
  if (n_in >= 60) {
    for (int i = 0; i < 53; ++i) P[i] = (const float*)d_in[7 + i];
  } else {
    const float* base = (const float*)d_in[7];
    size_t off = 0; int idx = 0;
    auto take = [&](size_t n) { P[idx++] = base + off; off += n; };
    take(10); take(2560);                       // head_b, head_w
    for (int l = 0; l < 3; ++l) {
      size_t Kin = (l == 0) ? 128 : 256;
      for (int r = 0; r < 8; ++r) take(256);    // att_dst x4, att_src x4
      take(256); take(65536);                   // k_b, k_w
      take(256); take(Kin * 256);               // author b,w
      take(256); take(Kin * 256);               // paper  b,w
      take(256); take(Kin * 256);               // term   b,w
      take(256);                                // q
    }
  }
  const float* head_b = P[0];
  const float* head_w = P[1];

  // ---------------- workspace carve-up (f32) ----------------
  float* ws = (float*)d_ws;
  size_t off = 0;
  auto alloc = [&](size_t n) { float* p = ws + off; off += n; return p; };
  float* hbuf[3];
  hbuf[0] = alloc((size_t)NA * HID);            // projected author
  hbuf[1] = alloc((size_t)NP * HID);            // projected paper
  hbuf[2] = alloc((size_t)NT * HID);            // projected term
  float* aggAP = alloc((size_t)NP * HID);       // dst=paper (relation author->paper)
  float* aggPA = alloc((size_t)NA * HID);       // dst=author
  float* aggPT = alloc((size_t)NT * HID);       // dst=term
  float* aggTP = alloc((size_t)NP * HID);       // dst=paper (relation term->paper)
  float* asrc  = alloc((size_t)NP * NHEAD);
  float* adst  = alloc((size_t)NP * NHEAD);
  float* denom = alloc((size_t)NP * NHEAD);
  unsigned* segmax = (unsigned*)alloc((size_t)NP * NHEAD);
  float* ebuf  = alloc((size_t)E * NHEAD);
  float* colsum = alloc(2 * HID);
  float* attn   = alloc(8);

  const int rsrc[4] = {0, 1, 1, 2};             // AP, PA, PT, TP source types
  const int rdst[4] = {1, 0, 2, 1};             // destination types
  float* ragg[4]    = {aggAP, aggPA, aggPT, aggTP};
  const int Ncnt[3] = {NA, NP, NT};

  auto gemm_blocks = [](int M) {                // 8 row tiles per block x 4 col groups
    int rowTiles = M >> 4;
    return ((rowTiles + 7) >> 3) * (HID >> 6);
  };

  const float* x[3] = {x0[0], x0[1], x0[2]};
  for (int l = 0; l < 3; ++l) {
    const float* const* LP = &P[2 + 17 * l];
    const float* att_dst[4] = {LP[0], LP[1], LP[2], LP[3]};
    const float* att_src[4] = {LP[4], LP[5], LP[6], LP[7]};
    const float* k_b = LP[8];
    const float* k_w = LP[9];
    const float* proj_b[3] = {LP[10], LP[12], LP[14]};
    const float* proj_w[3] = {LP[11], LP[13], LP[15]};
    const float* qv = LP[16];
    const int K = (l == 0) ? 128 : 256;

    // 1) per-node-type projections (WMMA f16)
    for (int t = 0; t < 3; ++t) {
      int M = Ncnt[t];
      gemm_wmma_f16<<<gemm_blocks(M), 256, 0, stream>>>(x[t], proj_w[t], proj_b[t],
                                                        hbuf[t], nullptr, M, K, HID, 0);
    }

    // 2) zero aggregation buffers (stream order: projections already consumed x aliases)
    hipMemsetAsync(aggAP, 0, (size_t)NP * HID * 4, stream);
    hipMemsetAsync(aggPA, 0, (size_t)NA * HID * 4, stream);
    hipMemsetAsync(aggPT, 0, (size_t)NT * HID * 4, stream);
    hipMemsetAsync(aggTP, 0, (size_t)NP * HID * 4, stream);

    // 3) per-relation GAT message passing with segment softmax
    for (int r = 0; r < 4; ++r) {
      const float* hs = hbuf[rsrc[r]]; int Ns = Ncnt[rsrc[r]];
      const float* hd = hbuf[rdst[r]]; int Nd = Ncnt[rdst[r]];
      att_reduce<<<(Ns * NHEAD + 255) / 256, 256, 0, stream>>>(hs, att_src[r], asrc, Ns);
      att_reduce<<<(Nd * NHEAD + 255) / 256, 256, 0, stream>>>(hd, att_dst[r], adst, Nd);
      hipMemsetAsync(segmax, 0, (size_t)Nd * NHEAD * 4, stream);
      hipMemsetAsync(denom,  0, (size_t)Nd * NHEAD * 4, stream);
      edge_logits<<<(E * NHEAD + 255) / 256, 256, 0, stream>>>(ei[r], asrc, adst, ebuf, segmax, E);
      edge_exp<<<(E * NHEAD + 255) / 256, 256, 0, stream>>>(ei[r], ebuf, segmax, denom, E);
      long long tot = (long long)E * HID;
      edge_msg<<<(int)((tot + 255) / 256), 256, 0, stream>>>(ei[r], ebuf, denom, hs, ragg[r], E);
    }

    // 4) relu on per-relation aggregations
    relu_ip<<<(NP * HID + 255) / 256, 256, 0, stream>>>(aggAP, NP * HID);
    relu_ip<<<(NA * HID + 255) / 256, 256, 0, stream>>>(aggPA, NA * HID);
    relu_ip<<<(NT * HID + 255) / 256, 256, 0, stream>>>(aggPT, NT * HID);
    relu_ip<<<(NP * HID + 255) / 256, 256, 0, stream>>>(aggTP, NP * HID);

    // 5) semantic attention: only paper has 2 relations (others: softmax(1 score)==1)
    hipMemsetAsync(colsum, 0, 2 * HID * 4, stream);
    gemm_wmma_f16<<<gemm_blocks(NP), 256, 0, stream>>>(aggAP, k_w, k_b, nullptr, colsum,
                                                       NP, HID, HID, 1);
    gemm_wmma_f16<<<gemm_blocks(NP), 256, 0, stream>>>(aggTP, k_w, k_b, nullptr, colsum + HID,
                                                       NP, HID, HID, 1);
    semantic_score<<<1, 1, 0, stream>>>(colsum, qv, 1.0f / NP, attn);
    combine2<<<(NP * HID + 255) / 256, 256, 0, stream>>>(aggAP, aggTP, attn, NP * HID);

    // next-layer inputs (aliases into agg buffers; consumed before next memset)
    x[0] = aggPA; x[1] = aggAP; x[2] = aggPT;
  }

  head_gemm<<<(NA * 10 + 255) / 256, 256, 0, stream>>>(x[0], head_w, head_b,
                                                       (float*)d_out, NA);
}
